// MaskedEncoderBlock_80496277061844
// MI455X (gfx1250) — matrix operand order repaired and checked by reading, not compiled, not run on hardware
//
#include <hip/hip_runtime.h>
#include <hip/hip_bf16.h>

// ---------------------------------------------------------------------------
// MaskedEncoderBlock for MI455X (gfx1250): bf16 WMMA GEMMs + flash attention.
// B=2, T=2048, C=2048, H=16, d=128.  Heavy math on v_wmma_f32_16x16x32_bf16;
// GEMM tiles staged into LDS by the Tensor Data Mover (tensor_load_to_lds)
// with double buffering + s_wait_tensorcnt.
// ---------------------------------------------------------------------------

typedef __attribute__((ext_vector_type(16))) __bf16 v16bf;
typedef __attribute__((ext_vector_type(8)))  __bf16 v8bf;
typedef __attribute__((ext_vector_type(8)))  float  v8f;
typedef __attribute__((ext_vector_type(4)))  unsigned short us4;
typedef __attribute__((ext_vector_type(4)))  unsigned int   u32x4;
typedef __attribute__((ext_vector_type(8)))  int            i32x8;
typedef __attribute__((ext_vector_type(4)))  int            i32x4;

#define CAT16(lo, hi) __builtin_shufflevector(lo, hi, 0,1,2,3,4,5,6,7,8,9,10,11,12,13,14,15)

__device__ __forceinline__ unsigned short f2bf(float f) {
    unsigned u = __float_as_uint(f);
    u += 0x7FFFu + ((u >> 16) & 1u);           // round to nearest even
    return (unsigned short)(u >> 16);
}

// ---------------------------------------------------------------------------
// TDM: issue one 2D tile DMA (global -> LDS), bf16 (2-byte) elements.
// D# per CDNA5 ISA ch.8: group0 = {count, lds_addr, global_addr, type=2},
// group1 = {data_size=1(2B), tensor_dim0/1, tile_dim0/1, dim0_stride}.
// ---------------------------------------------------------------------------
__device__ __forceinline__ void tdm_load_2d(const void* gsrc, unsigned lds_byte_off,
                                            unsigned dim0, unsigned dim1,
                                            unsigned stride0,
                                            unsigned tile0, unsigned tile1)
{
    unsigned long long ga = (unsigned long long)(size_t)gsrc;
    u32x4 g0;
    g0[0] = 1u;                                              // count=1, user descriptor
    g0[1] = lds_byte_off;                                    // lds_addr  [63:32]
    g0[2] = (unsigned)(ga & 0xFFFFFFFFull);                  // global_addr[31:0]
    g0[3] = (unsigned)((ga >> 32) & 0x1FFFFFFull)            // global_addr[56:32]
          | (2u << 30);                                      // type = 2 ("image")
    i32x8 g1;
    g1[0] = (int)(1u << 16);                                 // data_size=1 -> 2 bytes
    g1[1] = (int)((dim0 & 0xFFFFu) << 16);                   // tensor_dim0[15:0] @ [63:48]
    g1[2] = (int)(((dim0 >> 16) & 0xFFFFu)                   // tensor_dim0[31:16]
          |       ((dim1 & 0xFFFFu) << 16));                 // tensor_dim1[15:0]
    g1[3] = (int)(((dim1 >> 16) & 0xFFFFu)                   // tensor_dim1[31:16]
          |       ((tile0 & 0xFFFFu) << 16));                // tile_dim0 @ [127:112]
    g1[4] = (int)(tile1 & 0xFFFFu);                          // tile_dim1; tile_dim2=0
    g1[5] = (int)stride0;                                    // tensor_dim0_stride[31:0]
    g1[6] = 0;                                               // stride0[47:32], dim1_stride lo
    g1[7] = 0;
    i32x4 z4 = {0, 0, 0, 0};
#if defined(__clang_major__) && (__clang_major__ >= 23)
    i32x8 z8 = {};
    __builtin_amdgcn_tensor_load_to_lds(g0, g1, z4, z4, z8, 0);
#else
    __builtin_amdgcn_tensor_load_to_lds(g0, g1, z4, z4, 0);
#endif
}

// ---------------------------------------------------------------------------
// f32 -> bf16 conversion (vectorized x4)
// ---------------------------------------------------------------------------
__global__ __launch_bounds__(256)
void cvt_bf16(const float* __restrict__ in, unsigned short* __restrict__ out, int n) {
    int i = (blockIdx.x * 256 + threadIdx.x) * 4;
    if (i + 3 < n) {
        float4 v = *(const float4*)(in + i);
        us4 r;
        r.x = f2bf(v.x); r.y = f2bf(v.y); r.z = f2bf(v.z); r.w = f2bf(v.w);
        *(us4*)(out + i) = r;
    }
}

// ---------------------------------------------------------------------------
// GEMM: out[M,N] = A[M,K](bf16) @ W[N,K]^T(bf16) + bias[N]
// macro tile 128x64, BK=32; 256 threads = 8 waves, each wave 2x2 WMMA tiles.
// LDS double-buffered; tiles fetched by TDM issued from wave 0.
// mode 0: bf16 output, mode 1: f32 output.
// ---------------------------------------------------------------------------
#define GM 128
#define GN 64
#define GK 32

__global__ __launch_bounds__(256)
void gemm_bf16(const unsigned short* __restrict__ A,
               const unsigned short* __restrict__ W,
               const float* __restrict__ bias,
               unsigned short* __restrict__ outB,
               float* __restrict__ outF,
               int M, int N, int K, int mode)
{
    __shared__ __attribute__((aligned(16))) unsigned short As[2][GM * GK];  // 2 x 8 KB
    __shared__ __attribute__((aligned(16))) unsigned short Bs[2][GN * GK];  // 2 x 4 KB

    const int tid  = threadIdx.x;
    const int lane = tid & 31;
    const int wv   = tid >> 5;          // 0..7
    const int wm   = wv & 3;            // 4 waves along M (32 rows each)
    const int wn   = wv >> 2;           // 2 waves along N (32 cols each)
    const int hl   = lane >> 4;         // lane half (wave32)
    const int l16  = lane & 15;

    const int rowBase = blockIdx.x * GM;
    const int colBase = blockIdx.y * GN;

    const unsigned ldsA0 = (unsigned)(size_t)&As[0][0];
    const unsigned ldsA1 = (unsigned)(size_t)&As[1][0];
    const unsigned ldsB0 = (unsigned)(size_t)&Bs[0][0];
    const unsigned ldsB1 = (unsigned)(size_t)&Bs[1][0];

    v8f acc[2][2] = {};

    // ---- prologue: DMA tile 0 into buffer 0 ----
    if (wv == 0) {
        tdm_load_2d(A + (size_t)rowBase * K, ldsA0, K, M, K, GK, GM);
        tdm_load_2d(W + (size_t)colBase * K, ldsB0, K, N, K, GK, GN);
    }

    int cur = 0;
    for (int k0 = 0; k0 < K; k0 += GK) {
        __syncthreads();        // all waves done reading buf[cur^1] (previous compute)
        if (wv == 0) {
            if (k0 + GK < K) {
                // overlap: DMA next tile into the other buffer, then wait for current
                tdm_load_2d(A + (size_t)rowBase * K + k0 + GK,
                            cur ? ldsA0 : ldsA1, K, M, K, GK, GM);
                tdm_load_2d(W + (size_t)colBase * K + k0 + GK,
                            cur ? ldsB0 : ldsB1, K, N, K, GK, GN);
                __builtin_amdgcn_s_wait_tensorcnt(2);   // only the 2 new DMAs in flight
            } else {
                __builtin_amdgcn_s_wait_tensorcnt(0);
            }
        }
        __syncthreads();        // current tile visible to all waves

        // ---- fragments (ISA 7.12.2 layouts) ----
        v16bf afrag[2], bfrag[2];
        const int kb = hl * 8;  // A: lanes 0-15 -> k 0..7/16..23, lanes 16-31 -> k 8..15/24..31
#pragma unroll
        for (int mt = 0; mt < 2; ++mt) {
            const unsigned short* p = &As[cur][(wm * 32 + mt * 16 + l16) * GK + kb];
            v8bf lo = *(const v8bf*)p;
            v8bf hi = *(const v8bf*)(p + 16);
            afrag[mt] = CAT16(lo, hi);
        }
#pragma unroll
        for (int nt = 0; nt < 2; ++nt) {
            const unsigned short* p = &Bs[cur][(wn * 32 + nt * 16 + l16) * GK + hl * 8];
            v8bf lo = *(const v8bf*)p;
            v8bf hi = *(const v8bf*)(p + 16);
            bfrag[nt] = CAT16(lo, hi);
        }
#pragma unroll
        for (int mt = 0; mt < 2; ++mt)
#pragma unroll
            for (int nt = 0; nt < 2; ++nt)
                acc[mt][nt] = __builtin_amdgcn_wmma_f32_16x16x32_bf16(
                    false, afrag[mt], false, bfrag[nt], (short)0, acc[mt][nt], false, false);
        cur ^= 1;
    }

    // ---- epilogue: bias + store (C layout: row = i + 8*hl, col = l16) ----
#pragma unroll
    for (int mt = 0; mt < 2; ++mt)
#pragma unroll
        for (int nt = 0; nt < 2; ++nt) {
            int cidx = colBase + wn * 32 + nt * 16 + l16;
            float bv = bias[cidx];
#pragma unroll
            for (int i = 0; i < 8; ++i) {
                int ridx = rowBase + wm * 32 + mt * 16 + i + hl * 8;
                float v = acc[mt][nt][i] + bv;
                if (mode == 0) outB[(size_t)ridx * N + cidx] = f2bf(v);
                else           outF[(size_t)ridx * N + cidx] = v;
            }
        }
}

// ---------------------------------------------------------------------------
// Causal flash attention.  Grid: (T/128, B*H).  256 thr = 8 waves x 16 q rows.
// Q/K/V are bf16 [B*T, C] with head h at channels h*128..h*128+127.
// ---------------------------------------------------------------------------
#define SEQ   2048
#define CCH   2048
#define NHEAD 16
#define DHEAD 128
#define ABS   32      // keys per step

__global__ __launch_bounds__(256)
void attn_flash(const unsigned short* __restrict__ Qm,
                const unsigned short* __restrict__ Km,
                const unsigned short* __restrict__ Vm,
                unsigned short* __restrict__ Ym)
{
    __shared__ __attribute__((aligned(16))) unsigned short Ks[ABS * DHEAD];     // 8 KB  [s][d]
    __shared__ __attribute__((aligned(16))) unsigned short Vt[DHEAD * ABS];     // 8 KB  [d][s]
    __shared__ __attribute__((aligned(16))) unsigned short Ps[8 * 16 * ABS];    // 8 KB  per-wave P

    const int tid  = threadIdx.x;
    const int lane = tid & 31;
    const int wv   = tid >> 5;
    const int hl   = lane >> 4;
    const int l16  = lane & 15;

    const int bh = blockIdx.y;
    const int b  = bh / NHEAD;
    const int h  = bh % NHEAD;
    const int qbase = blockIdx.x * 128;
    const int qrow0 = qbase + wv * 16;                 // wave's first query row
    const size_t hdOff = (size_t)h * DHEAD;

    // ---- Q fragments held in registers: 4 k-chunks of 32 over d=128 ----
    v16bf qa[4];
#pragma unroll
    for (int c = 0; c < 4; ++c) {
        const unsigned short* p = Qm + ((size_t)(b * SEQ + qrow0 + l16)) * CCH + hdOff + c * 32 + hl * 8;
        v8bf lo = *(const v8bf*)p;
        v8bf hi = *(const v8bf*)(p + 16);
        qa[c] = CAT16(lo, hi);
    }

    v8f   o[8] = {};
    float mrun[8], lrun[8];
#pragma unroll
    for (int i = 0; i < 8; ++i) { mrun[i] = -3.0e38f; lrun[i] = 0.0f; }

    const float scale = 0.08838834764831845f;          // 1/sqrt(128)
    const int smax = qbase + 128;                      // causal bound for this WG

    for (int sb = 0; sb < smax; sb += ABS) {
        // ---- cooperative K / V^T staging ----
        {
            int r = tid >> 3;                          // 0..31 key row
            int c = (tid & 7) * 16;                    // d chunk
            const unsigned short* ksrc = Km + ((size_t)(b * SEQ + sb + r)) * CCH + hdOff + c;
            *(uint4*)&Ks[r * DHEAD + c]     = *(const uint4*)ksrc;
            *(uint4*)&Ks[r * DHEAD + c + 8] = *(const uint4*)(ksrc + 8);

            const unsigned short* vsrc = Vm + ((size_t)(b * SEQ + sb + r)) * CCH + hdOff + c;
            unsigned short tmp[16];
            *(uint4*)&tmp[0] = *(const uint4*)vsrc;
            *(uint4*)&tmp[8] = *(const uint4*)(vsrc + 8);
#pragma unroll
            for (int j = 0; j < 16; ++j) Vt[(c + j) * ABS + r] = tmp[j];
        }
        __syncthreads();

        if (sb <= qrow0 + 15) {                        // skip fully-masked blocks
            // ---- S = Q K^T : two 16x16 score tiles, K-dim = 128 in 4 WMMAs each ----
            v8f s[2] = {};
#pragma unroll
            for (int st = 0; st < 2; ++st)
#pragma unroll
                for (int c = 0; c < 4; ++c) {
                    const unsigned short* p = &Ks[(st * 16 + l16) * DHEAD + c * 32 + hl * 8];
                    v8bf lo = *(const v8bf*)p;
                    v8bf hi = *(const v8bf*)(p + 16);
                    v16bf bf = CAT16(lo, hi);
                    s[st] = __builtin_amdgcn_wmma_f32_16x16x32_bf16(
                        false, qa[c], false, bf, (short)0, s[st], false, false);
                }

            // ---- online softmax over 32 keys ----
            const int sg0 = sb + l16;
            const int sg1 = sb + 16 + l16;
#pragma unroll
            for (int i = 0; i < 8; ++i) {
                int qg = qrow0 + i + hl * 8;
                float a0 = s[0][i] * scale;
                float a1 = s[1][i] * scale;
                if (sg0 > qg) a0 = -1.0e30f;
                if (sg1 > qg) a1 = -1.0e30f;
                float mx = fmaxf(a0, a1);
#pragma unroll
                for (int off = 1; off < 16; off <<= 1)
                    mx = fmaxf(mx, __shfl_xor(mx, off, 32));
                float mnew = fmaxf(mrun[i], mx);
                float p0 = __expf(a0 - mnew);
                float p1 = __expf(a1 - mnew);
                float rs = p0 + p1;
#pragma unroll
                for (int off = 1; off < 16; off <<= 1)
                    rs += __shfl_xor(rs, off, 32);
                float alpha = __expf(mrun[i] - mnew);
                lrun[i] = lrun[i] * alpha + rs;
                mrun[i] = mnew;
#pragma unroll
                for (int dt = 0; dt < 8; ++dt) o[dt][i] *= alpha;
                // write P row to this wave's LDS buffer (C layout -> [row][key])
                int prow = i + hl * 8;
                Ps[wv * 512 + prow * ABS + l16]      = f2bf(p0);
                Ps[wv * 512 + prow * ABS + 16 + l16] = f2bf(p1);
            }

            // ---- O += P(16x32) @ V(32x128) : 8 WMMAs ----
            const unsigned short* pp = &Ps[wv * 512 + l16 * ABS + hl * 8];
            v8bf plo = *(const v8bf*)pp;
            v8bf phi = *(const v8bf*)(pp + 16);
            v16bf pa = CAT16(plo, phi);
#pragma unroll
            for (int dt = 0; dt < 8; ++dt) {
                const unsigned short* vp = &Vt[(dt * 16 + l16) * ABS + hl * 8];
                v8bf lo = *(const v8bf*)vp;
                v8bf hi = *(const v8bf*)(vp + 16);
                v16bf vb = CAT16(lo, hi);
                o[dt] = __builtin_amdgcn_wmma_f32_16x16x32_bf16(
                    false, pa, false, vb, (short)0, o[dt], false, false);
            }
        }
        __syncthreads();
    }

    // ---- normalize and store Y (bf16) ----
#pragma unroll
    for (int i = 0; i < 8; ++i) {
        float inv = 1.0f / lrun[i];
        int row = qrow0 + i + hl * 8;
        size_t base = ((size_t)(b * SEQ + row)) * CCH + hdOff;
#pragma unroll
        for (int dt = 0; dt < 8; ++dt)
            Ym[base + dt * 16 + l16] = f2bf(o[dt][i] * inv);
    }
}

// ---------------------------------------------------------------------------
// Row LayerNorm + residual:  out = LN(z * mask) * g + b + resid
// One 256-thread block per row of 2048; optional bf16 copy of the output.
// ---------------------------------------------------------------------------
__global__ __launch_bounds__(256)
void ln_residual(const float* __restrict__ z,
                 const float* __restrict__ resid,
                 const float* __restrict__ mask,      // per-row scalar or nullptr
                 const float* __restrict__ gamma,
                 const float* __restrict__ beta,
                 float* __restrict__ out,
                 unsigned short* __restrict__ outb,   // nullable
                 int C)
{
    __shared__ float red[256];
    const int row = blockIdx.x;
    const int tid = threadIdx.x;
    const float mrow = mask ? mask[row] : 1.0f;
    const float* zr = z + (size_t)row * C;

    float vals[8];
    float s = 0.0f;
#pragma unroll
    for (int j = 0; j < 8; ++j) {
        float v = zr[tid + j * 256] * mrow;
        vals[j] = v;
        s += v;
    }
    red[tid] = s; __syncthreads();
    for (int st = 128; st > 0; st >>= 1) {
        if (tid < st) red[tid] += red[tid + st];
        __syncthreads();
    }
    float mu = red[0] / C;
    __syncthreads();

    float vs = 0.0f;
#pragma unroll
    for (int j = 0; j < 8; ++j) {
        float d = vals[j] - mu;
        vs += d * d;
    }
    red[tid] = vs; __syncthreads();
    for (int st = 128; st > 0; st >>= 1) {
        if (tid < st) red[tid] += red[tid + st];
        __syncthreads();
    }
    float rstd = rsqrtf(red[0] / C + 1e-6f);

#pragma unroll
    for (int j = 0; j < 8; ++j) {
        int c = tid + j * 256;
        float y = (vals[j] - mu) * rstd * gamma[c] + beta[c] + resid[(size_t)row * C + c];
        out[(size_t)row * C + c] = y;
        if (outb) outb[(size_t)row * C + c] = f2bf(y);
    }
}

// ---------------------------------------------------------------------------
// Launch
// ---------------------------------------------------------------------------
extern "C" void kernel_launch(void* const* d_in, const int* in_sizes, int n_in,
                              void* d_out, int out_size, void* d_ws, size_t ws_size,
                              hipStream_t stream) {
    (void)in_sizes; (void)n_in; (void)out_size; (void)ws_size;
    const int Mr = 2 * SEQ;     // 4096 rows
    const int C  = CCH;         // 2048

    const float* x    = (const float*)d_in[0];
    const float* mask = (const float*)d_in[1];
    const float* Wq   = (const float*)d_in[2];
    const float* bq   = (const float*)d_in[3];
    const float* Wk   = (const float*)d_in[4];
    const float* bk   = (const float*)d_in[5];
    const float* Wv   = (const float*)d_in[6];
    const float* bv   = (const float*)d_in[7];
    const float* Wp   = (const float*)d_in[8];
    const float* bp   = (const float*)d_in[9];
    const float* Wfc  = (const float*)d_in[10];
    const float* bfc  = (const float*)d_in[11];
    const float* ln_g = (const float*)d_in[12];
    const float* ln_b = (const float*)d_in[13];

    unsigned char* ws = (unsigned char*)d_ws;
    const size_t MiB = 1024ull * 1024ull;
    unsigned short* Wqb  = (unsigned short*)(ws + 0 * MiB);
    unsigned short* Wkb  = (unsigned short*)(ws + 8 * MiB);
    unsigned short* Wvb  = (unsigned short*)(ws + 16 * MiB);
    unsigned short* Wpb  = (unsigned short*)(ws + 24 * MiB);
    unsigned short* Wfcb = (unsigned short*)(ws + 32 * MiB);
    unsigned short* Xb   = (unsigned short*)(ws + 40 * MiB);   // 16 MiB; reused as Y
    unsigned short* Qb   = (unsigned short*)(ws + 56 * MiB);
    unsigned short* Kb   = (unsigned short*)(ws + 72 * MiB);
    unsigned short* Vb   = (unsigned short*)(ws + 88 * MiB);
    unsigned short* Yb   = Xb;                                  // reuse (Xb dead after QKV)
    float*          F1   = (float*)(ws + 104 * MiB);            // 32 MiB GEMM f32 out (yp, then z)
    float*          X1   = (float*)(ws + 136 * MiB);            // 32 MiB
    unsigned short* X1b  = (unsigned short*)(ws + 168 * MiB);   // 16 MiB

    const int nW = C * C;           // 4M
    const int nX = Mr * C;          // 8M
    const int cvtB = 256 * 4;
    cvt_bf16<<<nW / cvtB, 256, 0, stream>>>(Wq,  Wqb,  nW);
    cvt_bf16<<<nW / cvtB, 256, 0, stream>>>(Wk,  Wkb,  nW);
    cvt_bf16<<<nW / cvtB, 256, 0, stream>>>(Wv,  Wvb,  nW);
    cvt_bf16<<<nW / cvtB, 256, 0, stream>>>(Wp,  Wpb,  nW);
    cvt_bf16<<<nW / cvtB, 256, 0, stream>>>(Wfc, Wfcb, nW);
    cvt_bf16<<<nX / cvtB, 256, 0, stream>>>(x,   Xb,   nX);

    dim3 gg(Mr / GM, C / GN);   // (32, 32)
    gemm_bf16<<<gg, 256, 0, stream>>>(Xb, Wqb, bq, Qb, nullptr, Mr, C, C, 0);
    gemm_bf16<<<gg, 256, 0, stream>>>(Xb, Wkb, bk, Kb, nullptr, Mr, C, C, 0);
    gemm_bf16<<<gg, 256, 0, stream>>>(Xb, Wvb, bv, Vb, nullptr, Mr, C, C, 0);

    attn_flash<<<dim3(SEQ / 128, 2 * NHEAD), 256, 0, stream>>>(Qb, Kb, Vb, Yb);

    gemm_bf16<<<gg, 256, 0, stream>>>(Yb, Wpb, bp, nullptr, F1, Mr, C, C, 1);
    ln_residual<<<Mr, 256, 0, stream>>>(F1, x, mask, ln_g, ln_b, X1, X1b, C);

    gemm_bf16<<<gg, 256, 0, stream>>>(X1b, Wfcb, bfc, nullptr, F1, Mr, C, C, 1);
    ln_residual<<<Mr, 256, 0, stream>>>(F1, X1, nullptr, ln_g, ln_b, (float*)d_out, nullptr, C);
}
